// VQMemory_32839319945804
// MI455X (gfx1250) — matrix operand order repaired and checked by reading, not compiled, not run on hardware
//
#include <hip/hip_runtime.h>
#include <stdint.h>

// ---------------- problem dims ----------------
#define M_DIM 64
#define B_DIM 16
#define T_DIM 2048
#define C_DIM 1024
#define H_DIM 16
#define HD    64
#define G_DIM 2
#define V_DIM 320
#define GV    640
#define D_DIM 512

// ---------------- types ----------------
typedef __attribute__((ext_vector_type(16))) __bf16 v16bf;
typedef __attribute__((ext_vector_type(8)))  __bf16 v8bf;
typedef __attribute__((ext_vector_type(8)))  float  v8f;

__device__ __forceinline__ unsigned short f2bf(float f) {
  union { float f; unsigned int u; } c; c.f = f;
  unsigned int u = c.u;
  unsigned int r = u + 0x7FFFu + ((u >> 16) & 1u);   // round-to-nearest-even
  return (unsigned short)(r >> 16);
}

__device__ __forceinline__ v8f wmma_bf16(v16bf a, v16bf b, v8f c) {
  // v_wmma_f32_16x16x32_bf16: (neg_a, A, neg_b, B, c_mod, C, reuse_a, reuse_b)
  return __builtin_amdgcn_wmma_f32_16x16x32_bf16(false, a, false, b, (short)0, c,
                                                 false, false);
}

// 16-element bf16 fragment from two 8-element (16B) chunks at +0 and +hi_off elems
__device__ __forceinline__ v16bf frag16(const unsigned short* p, int hi_off) {
  v8bf lo = *(const v8bf*)p;
  v8bf hi = *(const v8bf*)(p + hi_off);
  return __builtin_shufflevector(lo, hi, 0,1,2,3,4,5,6,7,8,9,10,11,12,13,14,15);
}

// A-fragment (16x32 bf16, CDNA5 wave32 layout) from a row pointer:
// elems 0..7 <- K = 8*half + (0..7) ; elems 8..15 <- K = 16 + 8*half + (0..7)
__device__ __forceinline__ v16bf load_afrag(const unsigned short* row, int c0, int half) {
  return frag16(row + c0 + 8 * half, 16);
}

// low 32 bits of a generic pointer to LDS == LDS byte address (aperture rule)
__device__ __forceinline__ unsigned lds_off32(const void* p) {
  return (unsigned)(unsigned long long)p;
}

// async global -> LDS copy of 16 bytes per lane (CDNA5; tracked by ASYNCcnt)
__device__ __forceinline__ void async_b128(unsigned lds_off, const void* gaddr) {
  asm volatile("global_load_async_to_lds_b128 %0, %1, off"
               :: "v"(lds_off), "v"(gaddr)
               : "memory");
}

// ---------------- f32 -> bf16 cast ----------------
__global__ void cast_f32_to_bf16(const float* __restrict__ in,
                                 unsigned short* __restrict__ out, int n) {
  for (int i = blockIdx.x * blockDim.x + threadIdx.x; i < n;
       i += gridDim.x * blockDim.x)
    out[i] = f2bf(in[i]);
}

// ---------------- generic GEMM: O[n,j] = ((sum_c A[n,c]*W[j,c]) + bias[j]) * alpha
// Block: 256 threads -> 128(n) x 64(j) tile; K-step 32; double-buffered async LDS staging.
enum { STORE_F32 = 0, STORE_BF16 = 1, STORE_BF16_VT = 2 };

#define APITCH 40   // LDS row pitch in bf16 elems (80 B) to spread banks
#define ABUF_BYTES (128 * APITCH * 2)
#define WBUF_BYTES (64 * APITCH * 2)

__global__ __launch_bounds__(256) void gemm_wmma_bf16(
    const unsigned short* __restrict__ A,  // [N,K] bf16
    const unsigned short* __restrict__ W,  // [J,K] bf16
    const float* __restrict__ bias,        // [J] f32
    float alpha, int N, int K, int J, int mode,
    float* __restrict__ Of,
    unsigned short* __restrict__ Ob) {
  __shared__ unsigned short Abuf[2][128 * APITCH];   // 20,480 B
  __shared__ unsigned short Wbuf[2][64 * APITCH];    // 10,240 B

  const int tilesN = (N + 127) >> 7;
  const int tn = blockIdx.x % tilesN, tj = blockIdx.x / tilesN;
  const int n0 = tn << 7, j0 = tj << 6;
  const int tid = threadIdx.x;
  const int lane = tid & 31, widx = tid >> 5;
  const int lrow = lane & 15, half = lane >> 4;
  const int ksteps = K >> 5;

  // per-thread async chunks: A gets 512 x 16B chunks (2/thread), W gets 256 (1/thread)
  const int arow0 = tid >> 2, apart = tid & 3;
  const int arow1 = (tid + 256) >> 2;
  const unsigned short* Ag0 = A + (size_t)(n0 + arow0) * K + apart * 8;
  const unsigned short* Ag1 = A + (size_t)(n0 + arow1) * K + apart * 8;
  const unsigned short* Wg  = W + (size_t)(j0 + arow0) * K + apart * 8;
  const unsigned al0 = lds_off32(&Abuf[0][arow0 * APITCH + apart * 8]);
  const unsigned al1 = lds_off32(&Abuf[0][arow1 * APITCH + apart * 8]);
  const unsigned wl0 = lds_off32(&Wbuf[0][arow0 * APITCH + apart * 8]);

  auto issue = [&](int s) {
    const int buf = s & 1;
    const int c0 = s << 5;
    async_b128(al0 + buf * ABUF_BYTES, Ag0 + c0);
    async_b128(al1 + buf * ABUF_BYTES, Ag1 + c0);
    async_b128(wl0 + buf * WBUF_BYTES, Wg + c0);
  };

  v8f acc[4] = {v8f{}, v8f{}, v8f{}, v8f{}};
  issue(0);
  for (int s = 0; s < ksteps; ++s) {
    const int cur = s & 1;
    if (s + 1 < ksteps) {
      issue(s + 1);
      asm volatile("s_wait_asynccnt 0x3" ::: "memory");  // buffer `cur` landed
    } else {
      asm volatile("s_wait_asynccnt 0x0" ::: "memory");
    }
    __syncthreads();

    // Batch all fragment loads before the WMMA chain so the backend can use
    // graduated s_wait_dscnt values instead of a full drain per WMMA.
    const unsigned short* Ab = &Abuf[cur][(widx * 16 + lrow) * APITCH];
    const unsigned short* Wb = &Wbuf[cur][lrow * APITCH + 8 * half];
    v16bf af = frag16(Ab + 8 * half, 16);
    v16bf b0 = frag16(Wb + 0 * 16 * APITCH, 16);
    v16bf b1 = frag16(Wb + 1 * 16 * APITCH, 16);
    v16bf b2 = frag16(Wb + 2 * 16 * APITCH, 16);
    v16bf b3 = frag16(Wb + 3 * 16 * APITCH, 16);
    acc[0] = wmma_bf16(af, b0, acc[0]);
    acc[1] = wmma_bf16(af, b1, acc[1]);
    acc[2] = wmma_bf16(af, b2, acc[2]);
    acc[3] = wmma_bf16(af, b3, acc[3]);
    __syncthreads();   // guard `cur` before next iteration's async writes
  }

#pragma unroll
  for (int jj = 0; jj < 4; ++jj) {
    const int j = j0 + jj * 16 + lrow;
    const float bj = bias ? bias[j] : 0.0f;
#pragma unroll
    for (int r = 0; r < 8; ++r) {
      const int n = n0 + widx * 16 + r + 8 * half;
      if (n >= N) continue;
      const float val = (acc[jj][r] + bj) * alpha;
      if (mode == STORE_F32) {
        Of[(size_t)n * J + j] = val;
      } else if (mode == STORE_BF16) {
        Ob[(size_t)n * J + j] = f2bf(val);
      } else {  // STORE_BF16_VT : n = t*B+b -> vT[j, b, t]  ([C,B,T])
        const int t = n / B_DIM, b = n % B_DIM;
        Ob[((size_t)j * B_DIM + b) * T_DIM + t] = f2bf(val);
      }
    }
  }
}

// ---------------- attention scores: S[b,h,m,t] = sum_d q[m,h*64+d]*k[(t*B+b),h*64+d]
__global__ void attn_scores_wmma(const unsigned short* __restrict__ q,   // [M,C] bf16 (pre-scaled)
                                 const unsigned short* __restrict__ k,   // [T*B,C] bf16
                                 float* __restrict__ scores) {           // [B,H,M,T] f32
  const int w = (int)((blockIdx.x * blockDim.x + threadIdx.x) >> 5);  // 32768 waves
  const int tt = w & 31;             // t tile of 64
  const int tm = (w >> 5) & 3;       // m tile of 16
  const int h  = (w >> 7) & 15;
  const int b  = w >> 11;
  const int lane = threadIdx.x & 31;
  const int lrow = lane & 15, half = lane >> 4;

  const unsigned short* Qrow = q + (size_t)(tm * 16 + lrow) * C_DIM + h * HD;
  v8f acc[4] = {v8f{}, v8f{}, v8f{}, v8f{}};

#pragma unroll
  for (int d0 = 0; d0 < HD; d0 += 32) {
    v16bf af = frag16(Qrow + d0 + 16 * half, 8);
#pragma unroll
    for (int jj = 0; jj < 4; ++jj) {
      const int t = tt * 64 + jj * 16 + lrow;
      const unsigned short* kp =
          k + ((size_t)t * B_DIM + b) * C_DIM + h * HD + d0 + 16 * half;
      v16bf bf = *(const v16bf*)kp;
      acc[jj] = wmma_bf16(af, bf, acc[jj]);
    }
  }

  const size_t base = (((size_t)b * H_DIM + h) * M_DIM + tm * 16) * T_DIM;
#pragma unroll
  for (int jj = 0; jj < 4; ++jj)
#pragma unroll
    for (int r = 0; r < 8; ++r)
      scores[base + (size_t)(r + 8 * half) * T_DIM + tt * 64 + jj * 16 + lrow] =
          acc[jj][r];
}

// ---------------- row softmax over T with padding mask; emit bf16 probs ----------------
__global__ void softmax_rows(const float* __restrict__ scores,
                             const unsigned char* __restrict__ mask,     // [B,T]
                             unsigned short* __restrict__ attn) {        // [B,H,M,T] bf16
  const int row = blockIdx.x;                 // b*H*M + h*M + m (1024 per b)
  const int b = row >> 10;
  const int tid = threadIdx.x, lane = tid & 31, wid = tid >> 5;
  const float* srow = scores + (size_t)row * T_DIM;
  unsigned short* arow = attn + (size_t)row * T_DIM;
  const unsigned char* mrow = mask + (size_t)b * T_DIM;
  __shared__ float sred[8];

  float vals[8];
  float mx = -3.0e38f;
#pragma unroll
  for (int i = 0; i < 8; ++i) {
    const int t = tid + (i << 8);
    float s = srow[t];
    if (mrow[t]) s = -3.0e38f;
    vals[i] = s;
    mx = fmaxf(mx, s);
  }
  for (int o = 16; o > 0; o >>= 1) mx = fmaxf(mx, __shfl_xor(mx, o, 32));
  if (lane == 0) sred[wid] = mx;
  __syncthreads();
  if (tid == 0) { float v = sred[0]; for (int i = 1; i < 8; ++i) v = fmaxf(v, sred[i]); sred[0] = v; }
  __syncthreads();
  mx = sred[0];
  __syncthreads();

  float sum = 0.0f;
#pragma unroll
  for (int i = 0; i < 8; ++i) { vals[i] = __expf(vals[i] - mx); sum += vals[i]; }
  for (int o = 16; o > 0; o >>= 1) sum += __shfl_xor(sum, o, 32);
  if (lane == 0) sred[wid] = sum;
  __syncthreads();
  if (tid == 0) { float v = 0; for (int i = 0; i < 8; ++i) v += sred[i]; sred[0] = v; }
  __syncthreads();
  const float inv = sred[0] > 0.0f ? 1.0f / sred[0] : 0.0f;
#pragma unroll
  for (int i = 0; i < 8; ++i) arow[tid + (i << 8)] = f2bf(vals[i] * inv);
}

// ---------------- ctx[b,h,m,d] = sum_t attn[b,h,m,t] * v[t,b,h*64+d] -> ctx2d[b*M+m, C]
__global__ void attn_ctx_wmma(const unsigned short* __restrict__ attn,  // [B,H,M,T] bf16
                              const unsigned short* __restrict__ vT,    // [C,B,T] bf16
                              unsigned short* __restrict__ ctx2d) {     // [B*M, C] bf16
  const int w = (int)((blockIdx.x * blockDim.x + threadIdx.x) >> 5);  // 1024 waves
  const int tm = w & 3;
  const int h  = (w >> 2) & 15;
  const int b  = w >> 6;
  const int lane = threadIdx.x & 31;
  const int lrow = lane & 15, half = lane >> 4;

  const unsigned short* Arow =
      attn + (((size_t)b * H_DIM + h) * M_DIM + tm * 16 + lrow) * T_DIM;
  v8f acc[4] = {v8f{}, v8f{}, v8f{}, v8f{}};

  for (int t0 = 0; t0 < T_DIM; t0 += 32) {
    v16bf af = frag16(Arow + t0 + 16 * half, 8);
#pragma unroll
    for (int jj = 0; jj < 4; ++jj) {
      const unsigned short* vp =
          vT + ((size_t)(h * HD + jj * 16 + lrow) * B_DIM + b) * T_DIM + t0 + 16 * half;
      v16bf bf = *(const v16bf*)vp;
      acc[jj] = wmma_bf16(af, bf, acc[jj]);
    }
  }

#pragma unroll
  for (int jj = 0; jj < 4; ++jj)
#pragma unroll
    for (int r = 0; r < 8; ++r)
      ctx2d[((size_t)b * M_DIM + tm * 16 + r + 8 * half) * C_DIM +
            h * HD + jj * 16 + lrow] = f2bf(acc[jj][r]);
}

// ---------------- VQ: softmax(logits) + hard gumbel argmax -> codebook gather ----------------
__global__ void vq_finalize(const float* __restrict__ logits,    // [B*M, 640]
                            const float* __restrict__ gumbel,    // [B*M, 640]
                            const float* __restrict__ codebook,  // [640, 512]
                            float* __restrict__ out,             // [M,B,1024]
                            float* __restrict__ vql) {           // [B*M,2,320]
  const int n = blockIdx.x >> 1, g = blockIdx.x & 1;
  const int tid = threadIdx.x, lane = tid & 31, wid = tid >> 5;
  const float* lrow = logits + (size_t)n * GV + g * V_DIM;
  const float* grow = gumbel + (size_t)n * GV + g * V_DIM;
  __shared__ float sred[8];
  __shared__ float swv[8];
  __shared__ int   swi[8];
  __shared__ int   sidx;

  const float a0 = lrow[tid];
  const bool has2 = tid < (V_DIM - 256);              // tid < 64
  const float a1 = has2 ? lrow[tid + 256] : -3.0e38f;

  // row max of logits
  float mx = fmaxf(a0, a1);
  for (int o = 16; o > 0; o >>= 1) mx = fmaxf(mx, __shfl_xor(mx, o, 32));
  if (lane == 0) sred[wid] = mx;
  __syncthreads();
  if (tid == 0) { float v = sred[0]; for (int i = 1; i < 8; ++i) v = fmaxf(v, sred[i]); sred[0] = v; }
  __syncthreads();
  mx = sred[0];
  __syncthreads();

  // softmax (vq_logits output)
  const float e0 = __expf(a0 - mx);
  const float e1 = has2 ? __expf(a1 - mx) : 0.0f;
  float s = e0 + e1;
  for (int o = 16; o > 0; o >>= 1) s += __shfl_xor(s, o, 32);
  if (lane == 0) sred[wid] = s;
  __syncthreads();
  if (tid == 0) { float v = 0; for (int i = 0; i < 8; ++i) v += sred[i]; sred[0] = v; }
  __syncthreads();
  const float inv = 1.0f / sred[0];
  float* vrow = vql + (size_t)n * GV + g * V_DIM;
  vrow[tid] = e0 * inv;
  if (has2) vrow[tid + 256] = e1 * inv;

  // argmax(logits + gumbel): forward value of hard gumbel-softmax ST (ties -> lowest idx)
  float ga = a0 + grow[tid]; int gi = tid;
  if (has2) {
    const float gb = lrow[tid + 256] + grow[tid + 256];
    if (gb > ga) { ga = gb; gi = tid + 256; }
  }
  for (int o = 16; o > 0; o >>= 1) {
    const float ov = __shfl_xor(ga, o, 32);
    const int   oi = __shfl_xor(gi, o, 32);
    if (ov > ga || (ov == ga && oi < gi)) { ga = ov; gi = oi; }
  }
  if (lane == 0) { swv[wid] = ga; swi[wid] = gi; }
  __syncthreads();
  if (tid == 0) {
    float bv = swv[0]; int bi = swi[0];
    for (int i = 1; i < 8; ++i)
      if (swv[i] > bv || (swv[i] == bv && swi[i] < bi)) { bv = swv[i]; bi = swi[i]; }
    sidx = bi;
  }
  __syncthreads();

  const int idx = sidx;
  const float* crow = codebook + ((size_t)g * V_DIM + idx) * D_DIM;
  const int b = n >> 6, m = n & 63;                     // n = b*M + m
  float* orow = out + ((size_t)m * B_DIM + b) * (G_DIM * D_DIM) + (size_t)g * D_DIM;
  orow[tid] = crow[tid];
  orow[tid + 256] = crow[tid + 256];
}

// ---------------- launch ----------------
extern "C" void kernel_launch(void* const* d_in, const int* in_sizes, int n_in,
                              void* d_out, int out_size, void* d_ws, size_t ws_size,
                              hipStream_t stream) {
  const float* x        = (const float*)d_in[0];
  const unsigned char* pmask = (const unsigned char*)d_in[1];
  const float* gumbel   = (const float*)d_in[2];
  const float* memory   = (const float*)d_in[3];
  const float* Wq = (const float*)d_in[4];  const float* bq = (const float*)d_in[5];
  const float* Wk = (const float*)d_in[6];  const float* bk = (const float*)d_in[7];
  const float* Wv = (const float*)d_in[8];  const float* bv = (const float*)d_in[9];
  const float* Wo = (const float*)d_in[10]; const float* bo = (const float*)d_in[11];
  const float* Wp = (const float*)d_in[12]; const float* bp = (const float*)d_in[13];
  const float* codebook = (const float*)d_in[14];

  // scratch layout (requires ws_size >= ~353 MB)
  char* ws = (char*)d_ws;
  const size_t SC_SCORES = 0;                                  // 134,217,728 (B*H*M*T f32)
  const size_t SC_K      = SC_SCORES + (size_t)134217728;      //  67,108,864 (T*B*C bf16)
  const size_t SC_VT     = SC_K      + (size_t)67108864;       //  67,108,864 (C*B*T bf16)
  const size_t SC_X      = SC_VT     + (size_t)67108864;       //  67,108,864 (x bf16; reused as attn)
  const size_t SC_WQ     = SC_X      + (size_t)67108864;
  const size_t SC_WK     = SC_WQ + 2097152;
  const size_t SC_WV     = SC_WK + 2097152;
  const size_t SC_WO     = SC_WV + 2097152;
  const size_t SC_WP     = SC_WO + 2097152;                    // 1,310,720 used
  const size_t SC_MEM    = SC_WP + 1310720;                    //   131,072
  const size_t SC_Q      = SC_MEM + 131072;                    //   131,072
  const size_t SC_CTX    = SC_Q  + 131072;                     // 2,097,152
  const size_t SC_H      = SC_CTX + 2097152;                   // 2,097,152
  const size_t SC_LOG    = SC_H  + 2097152;                    // 2,621,440

  float*          s_scores = (float*)(ws + SC_SCORES);
  unsigned short* s_k      = (unsigned short*)(ws + SC_K);
  unsigned short* s_vT     = (unsigned short*)(ws + SC_VT);
  unsigned short* s_x      = (unsigned short*)(ws + SC_X);
  unsigned short* s_attn   = (unsigned short*)(ws + SC_X);     // overlays x (dead after K/V)
  unsigned short* s_Wq     = (unsigned short*)(ws + SC_WQ);
  unsigned short* s_Wk     = (unsigned short*)(ws + SC_WK);
  unsigned short* s_Wv     = (unsigned short*)(ws + SC_WV);
  unsigned short* s_Wo     = (unsigned short*)(ws + SC_WO);
  unsigned short* s_Wp     = (unsigned short*)(ws + SC_WP);
  unsigned short* s_mem    = (unsigned short*)(ws + SC_MEM);
  unsigned short* s_q      = (unsigned short*)(ws + SC_Q);
  unsigned short* s_ctx    = (unsigned short*)(ws + SC_CTX);
  unsigned short* s_h      = (unsigned short*)(ws + SC_H);
  float*          s_logits = (float*)(ws + SC_LOG);

  // 1) bf16 casts
  cast_f32_to_bf16<<<2048, 256, 0, stream>>>(x,      s_x,   T_DIM * B_DIM * C_DIM);
  cast_f32_to_bf16<<<64,   256, 0, stream>>>(memory, s_mem, M_DIM * C_DIM);
  cast_f32_to_bf16<<<1024, 256, 0, stream>>>(Wq, s_Wq, C_DIM * C_DIM);
  cast_f32_to_bf16<<<1024, 256, 0, stream>>>(Wk, s_Wk, C_DIM * C_DIM);
  cast_f32_to_bf16<<<1024, 256, 0, stream>>>(Wv, s_Wv, C_DIM * C_DIM);
  cast_f32_to_bf16<<<1024, 256, 0, stream>>>(Wo, s_Wo, C_DIM * C_DIM);
  cast_f32_to_bf16<<<1024, 256, 0, stream>>>(Wp, s_Wp, GV * C_DIM);

  auto gemm_blocks = [](int N, int J) { return ((N + 127) / 128) * (J / 64); };

  // 2) Q = (mem @ Wq^T + bq) * hd^-0.5   (computed once; B-broadcast implicit)
  gemm_wmma_bf16<<<gemm_blocks(M_DIM, C_DIM), 256, 0, stream>>>(
      s_mem, s_Wq, bq, 0.125f, M_DIM, C_DIM, C_DIM, STORE_BF16, nullptr, s_q);
  // 3) K = x @ Wk^T + bk  (row-major [T*B, C])
  gemm_wmma_bf16<<<gemm_blocks(T_DIM * B_DIM, C_DIM), 256, 0, stream>>>(
      s_x, s_Wk, bk, 1.0f, T_DIM * B_DIM, C_DIM, C_DIM, STORE_BF16, nullptr, s_k);
  // 4) V = x @ Wv^T + bv, stored transposed [C, B, T] for the ctx WMMA B-fragments
  gemm_wmma_bf16<<<gemm_blocks(T_DIM * B_DIM, C_DIM), 256, 0, stream>>>(
      s_x, s_Wv, bv, 1.0f, T_DIM * B_DIM, C_DIM, C_DIM, STORE_BF16_VT, nullptr, s_vT);

  // 5) scores (fp32), 6) softmax -> bf16 attn (overlays x), 7) ctx
  attn_scores_wmma<<<4096, 256, 0, stream>>>(s_q, s_k, s_scores);
  softmax_rows<<<B_DIM * H_DIM * M_DIM, 256, 0, stream>>>(s_scores, pmask, s_attn);
  attn_ctx_wmma<<<128, 256, 0, stream>>>(s_attn, s_vT, s_ctx);

  // 8) h = ctx @ Wo^T + bo ; 9) logits = h @ Wp^T + bp (fp32 for softmax/argmax)
  gemm_wmma_bf16<<<gemm_blocks(B_DIM * M_DIM, C_DIM), 256, 0, stream>>>(
      s_ctx, s_Wo, bo, 1.0f, B_DIM * M_DIM, C_DIM, C_DIM, STORE_BF16, nullptr, s_h);
  gemm_wmma_bf16<<<gemm_blocks(B_DIM * M_DIM, GV), 256, 0, stream>>>(
      s_h, s_Wp, bp, 1.0f, B_DIM * M_DIM, C_DIM, GV, STORE_F32, s_logits, nullptr);

  // 10) VQ outputs: out [M,B,1024] then vq_logits [B*M,2,320]
  float* out0 = (float*)d_out;
  float* out1 = out0 + (size_t)M_DIM * B_DIM * G_DIM * D_DIM;  // +1,048,576
  vq_finalize<<<B_DIM * M_DIM * G_DIM, 256, 0, stream>>>(s_logits, gumbel, codebook,
                                                         out0, out1);
}
